// GraphAutoencoder_34230889349208
// MI455X (gfx1250) — matrix operand order repaired and checked by reading, not compiled, not run on hardware
//
#include <hip/hip_runtime.h>
#include <math.h>

// ---------------- problem constants (reference: N=10000, E=320000) ----------
#define GN 10000
#define GE 320000
// GE % 16 == 0 -> no partial edge tiles; EXEC stays all-ones around every
// WMMA (hard requirement, ISA 7.12). Decoder handles GN % 32 != 0 by
// clamping load rows and predicating only stores.

typedef float    v2f  __attribute__((ext_vector_type(2)));
typedef float    v8f  __attribute__((ext_vector_type(8)));
typedef _Float16 v16h __attribute__((ext_vector_type(16)));
typedef unsigned int u32x4 __attribute__((ext_vector_type(4)));
typedef int          i32x4 __attribute__((ext_vector_type(4)));
typedef int          i32x8 __attribute__((ext_vector_type(8)));

#if defined(__has_builtin)
#if __has_builtin(__builtin_amdgcn_wmma_f32_16x16x4_f32)
#define HAVE_WMMA_F32 1
#endif
#if __has_builtin(__builtin_amdgcn_tensor_load_to_lds)
#define HAVE_TDM 1
#endif
#endif

__device__ inline void wait_tensorcnt0() {
#if defined(__has_builtin) && __has_builtin(__builtin_amdgcn_s_wait_tensorcnt)
  __builtin_amdgcn_s_wait_tensorcnt(0);
#else
  asm volatile("s_wait_tensorcnt 0x0" ::: "memory");
#endif
}

#ifdef HAVE_TDM
// ---------------------------------------------------------------------------
// TDM: load a [rows x cols] fp32 tile (row stride row_stride elems) from
// global into LDS at byte offset lds_off.  Descriptor per ISA ch.8:
//   group0: count=1 | lds_addr | global_addr[56:0] | type=2
//   group1: data_size=2(4B) | tensor_dim0/1 | tile_dim0/1 | dim0_stride
//   groups 2/3: zero (2-D tile; tile_dim2=0 marks higher dims unused)
// ---------------------------------------------------------------------------
__device__ inline void tdm_load_2d_f32(unsigned lds_off, const void* gptr,
                                       unsigned rows, unsigned cols,
                                       unsigned row_stride) {
  const unsigned long long ga = (unsigned long long)gptr;
  u32x4 g0;
  g0[0] = 1u;                                  // count=1, no gather/restore
  g0[1] = lds_off;                             // lds_addr (bytes)
  g0[2] = (unsigned)(ga & 0xffffffffu);        // global_addr[31:0]
  g0[3] = (unsigned)((ga >> 32) & 0x1ffffffu)  // global_addr[56:32]
          | (2u << 30);                        // type = 2 ("image")
  i32x8 g1;
  g1[0] = (int)(2u << 16);                     // data_size=2 -> 4 bytes
  g1[1] = (int)((cols & 0xffffu) << 16);       // tensor_dim0[15:0]
  g1[2] = (int)(((cols >> 16) & 0xffffu) |     // tensor_dim0[31:16]
                ((rows & 0xffffu) << 16));     // tensor_dim1[15:0]
  g1[3] = (int)(((rows >> 16) & 0xffffu) |     // tensor_dim1[31:16]
                ((cols & 0xffffu) << 16));     // tile_dim0 = cols
  g1[4] = (int)(rows & 0xffffu);               // tile_dim1 = rows, tile_dim2=0
  g1[5] = (int)row_stride;                     // tensor_dim0_stride[31:0]
  g1[6] = 0;                                   // stride[47:32], dim1_stride lo
  g1[7] = 0;
  const i32x4 z4 = {0, 0, 0, 0};
#if __clang_major__ >= 23
  const i32x8 z8 = {0, 0, 0, 0, 0, 0, 0, 0};
  __builtin_amdgcn_tensor_load_to_lds(g0, g1, z4, z4, z8, 0);
#else
  __builtin_amdgcn_tensor_load_to_lds(g0, g1, z4, z4, 0);
#endif
}
#endif  // HAVE_TDM

// lds byte offset of a __shared__ object (flat LDS aperture: low 32 bits)
template <class T>
__device__ inline unsigned lds_offset(T* p) {
  return (unsigned)(unsigned long long)(void*)p;
}

// ---------------------------------------------------------------------------
// Generic 16x16 fp32 output tile:  acc = sum_k A(r,k) * B(k,c)
// One wave32 per tile; lane mapping per ISA 7.12.2 VGPR layout tables.
// ---------------------------------------------------------------------------
template <class LA, class LB>
__device__ inline v8f wmma_tile(int Kin, LA la, LB lb, int lane) {
  const int lhi = lane >> 4;
  const int llo = lane & 15;
  v8f acc = {};
#ifdef HAVE_WMMA_F32
  // V_WMMA_F32_16X16X4_F32.  A 16x4: row=llo, K = k0 + 2*lhi + j.
  for (int k0 = 0; k0 < Kin; k0 += 4) {
    const int ka = k0 + 2 * lhi;
    v2f a, b;
    a.x = la(llo, ka);
    a.y = la(llo, ka + 1);
    b.x = lb(ka, llo);
    b.y = lb(ka + 1, llo);
    acc = __builtin_amdgcn_wmma_f32_16x16x4_f32(
        false, a, false, b, (short)0, acc, false, false);
  }
#else
  for (int k0 = 0; k0 < Kin; k0 += 32) {
    v16h a, b;
#pragma unroll
    for (int h = 0; h < 8; ++h) {
      a[h]     = (_Float16)la(llo, k0 + 8 * lhi + h);
      a[h + 8] = (_Float16)la(llo, k0 + 16 + 8 * lhi + h);
    }
#pragma unroll
    for (int h = 0; h < 16; ++h)
      b[h] = (_Float16)lb(k0 + 16 * lhi + h, llo);
    acc = __builtin_amdgcn_wmma_f32_16x16x32_f16(
        false, a, false, b, (short)0, acc, false, false);
  }
#endif
  return acc;
}

// ---------------------------------------------------------------------------
// Y[r, c] = X[r, :Kin] @ W[:Kin, c] + b[c]      (one wave per 16x16 tile)
// 16-col weight panel staged once in LDS (coalesced) -> ds_load B fragments.
// ---------------------------------------------------------------------------
__global__ __launch_bounds__(32) void fc_kernel(
    const float* __restrict__ X, const float* __restrict__ W,
    const float* __restrict__ b, float* __restrict__ Y, int Kin, int Kout) {
  __shared__ float wtile[128 * 16];  // Kin <= 128
  const int lane = threadIdx.x;
  const int lhi = lane >> 4, llo = lane & 15;
  const int r0 = blockIdx.x * 16;
  const int c0 = blockIdx.y * 16;

  for (int idx = lane; idx < Kin * 16; idx += 32)
    wtile[idx] = W[(size_t)(idx >> 4) * Kout + c0 + (idx & 15)];
  __syncthreads();

  v8f acc = wmma_tile(
      Kin,
      [&](int r, int k) { return X[(size_t)(r0 + r) * Kin + k]; },
      [&](int k, int c) { return wtile[(k << 4) + c]; }, lane);

  const float bias = b[c0 + llo];
#pragma unroll
  for (int i = 0; i < 8; ++i)
    Y[(size_t)(r0 + i + 8 * lhi) * Kout + (c0 + llo)] = acc[i] + bias;
}

// ---------------------------------------------------------------------------
// Fused edge-message kernel.  One wave per block; each block owns ET=32
// consecutive 16-edge tiles (512 edges) so the TDM-staged We panel
// (64 x D fp32 in LDS) amortizes.  Per tile:
//   t   = h[src] + ef @ We + be     (ef tile TDM-staged, t in LDS)
//   msg = t @ Wm + bm
//   agg[dst] += msg                 (global_atomic_add_f32, L2-resident)
// ---------------------------------------------------------------------------
#define ET 32

template <int D>
__global__ __launch_bounds__(32) void msg_kernel(
    const float* __restrict__ ef, const int* __restrict__ src,
    const int* __restrict__ dst, const float* __restrict__ h,
    const float* __restrict__ We, const float* __restrict__ be,
    const float* __restrict__ Wm, const float* __restrict__ bm,
    float* __restrict__ agg) {
  __shared__ float we_s[64 * D];   // staged weight panel
  __shared__ float ef_s[16 * 64];  // staged edge-feature tile
  __shared__ float t[16 * D];
  __shared__ int sidx[16];
  __shared__ int didx[16];

  const int lane = threadIdx.x;
  const int lhi = lane >> 4, llo = lane & 15;

#ifdef HAVE_TDM
  // async tensor DMA: whole We panel -> LDS (64 rows x D cols, stride D)
  tdm_load_2d_f32(lds_offset(we_s), We, 64, (unsigned)D, (unsigned)D);
#else
  for (int idx = lane; idx < 64 * D; idx += 32)
    we_s[idx] = We[(size_t)(idx / D) * D + (idx % D)];
#endif

  for (int ti = 0; ti < ET; ++ti) {
    const int e0 = (blockIdx.x * ET + ti) * 16;
    __syncthreads();  // previous iteration done reading ef_s/t/idx

    if (lane < 16) {
      sidx[lane] = src[e0 + lane];
      didx[lane] = dst[e0 + lane];
    }
#ifdef HAVE_TDM
    tdm_load_2d_f32(lds_offset(ef_s), ef + (size_t)e0 * 64, 16, 64, 64);
#else
    for (int idx = lane; idx < 16 * 64; idx += 32)
      ef_s[idx] = ef[(size_t)(e0 + (idx >> 6)) * 64 + (idx & 63)];
#endif
    __syncthreads();  // sidx visible

    // gather h[src] rows into LDS tile (overlaps with the tensor DMA)
    for (int idx = lane; idx < 16 * D; idx += 32)
      t[idx] = h[(size_t)sidx[idx / D] * D + (idx % D)];
    wait_tensorcnt0();  // ef_s (and on first iter we_s) resident
    __syncthreads();

    // t += ef @ We + be   (K = 64)
    for (int ct = 0; ct < D / 16; ++ct) {
      const int c0 = ct * 16;
      v8f acc = wmma_tile(
          64,
          [&](int r, int k) { return ef_s[(r << 6) + k]; },
          [&](int k, int c) { return we_s[k * D + c0 + c]; }, lane);
      const float bias = be[c0 + llo];
#pragma unroll
      for (int i = 0; i < 8; ++i)
        t[(i + 8 * lhi) * D + c0 + llo] += acc[i] + bias;
    }
    __syncthreads();

    // msg = t @ Wm + bm; scatter-add into agg[dst]
    for (int ct = 0; ct < D / 16; ++ct) {
      const int c0 = ct * 16;
      v8f acc = wmma_tile(
          D,
          [&](int r, int k) { return t[r * D + k]; },
          [&](int k, int c) { return Wm[(size_t)k * D + c0 + c]; }, lane);
      const float bias = bm[c0 + llo];
#pragma unroll
      for (int i = 0; i < 8; ++i) {
        const int r = i + 8 * lhi;
        atomicAdd(&agg[(size_t)didx[r] * D + c0 + llo], acc[i] + bias);
      }
    }
  }
}

// ---------------------------------------------------------------------------
// Decoder: adj[m,n] = sigmoid(dot(z[m,:64], z[n,:64])), f16 WMMA, K=64.
// 32x32 register tile per wave (4 accumulators, 2 A + 2 B fragments per
// K-step) -> halves fragment traffic per FLOP vs 16x16 tiles.  Load rows
// are clamped (EXEC stays all-ones for WMMA); only stores are predicated.
// ---------------------------------------------------------------------------
__global__ __launch_bounds__(32) void adj_kernel(
    const _Float16* __restrict__ zh, float* __restrict__ adj) {
  const int lane = threadIdx.x;
  const int lhi = lane >> 4, llo = lane & 15;
  const int m0 = blockIdx.x * 32;
  const int n0 = blockIdx.y * 32;

  const int ra0 = (m0 + llo < GN) ? m0 + llo : GN - 1;        // A rows, tile 0
  const int ra1 = (m0 + 16 + llo < GN) ? m0 + 16 + llo : GN - 1;
  const int rb0 = (n0 + llo < GN) ? n0 + llo : GN - 1;        // B rows (cols)
  const int rb1 = (n0 + 16 + llo < GN) ? n0 + 16 + llo : GN - 1;

  v8f acc00 = {}, acc01 = {}, acc10 = {}, acc11 = {};
#pragma unroll
  for (int k0 = 0; k0 < 64; k0 += 32) {
    v16h a0, a1, b0, b1;
    // A fragment (16x32 f16): K = k0 + 16*lhi + h (contiguous, matches B)
    const _Float16* a0p = zh + (size_t)ra0 * 64 + k0 + 16 * lhi;
    const _Float16* a1p = zh + (size_t)ra1 * 64 + k0 + 16 * lhi;
#pragma unroll
    for (int hh = 0; hh < 8; ++hh) {
      a0[hh] = a0p[hh];
      a0[hh + 8] = a0p[8 + hh];
      a1[hh] = a1p[hh];
      a1[hh + 8] = a1p[8 + hh];
    }
    // B fragment (32x16): B[k][n] = z[n][k] -> 16 contiguous halfs per lane
    const _Float16* b0p = zh + (size_t)rb0 * 64 + k0 + 16 * lhi;
    const _Float16* b1p = zh + (size_t)rb1 * 64 + k0 + 16 * lhi;
#pragma unroll
    for (int hh = 0; hh < 16; ++hh) {
      b0[hh] = b0p[hh];
      b1[hh] = b1p[hh];
    }
    acc00 = __builtin_amdgcn_wmma_f32_16x16x32_f16(false, a0, false, b0,
                                                   (short)0, acc00, false, false);
    acc01 = __builtin_amdgcn_wmma_f32_16x16x32_f16(false, a0, false, b1,
                                                   (short)0, acc01, false, false);
    acc10 = __builtin_amdgcn_wmma_f32_16x16x32_f16(false, a1, false, b0,
                                                   (short)0, acc10, false, false);
    acc11 = __builtin_amdgcn_wmma_f32_16x16x32_f16(false, a1, false, b1,
                                                   (short)0, acc11, false, false);
  }

#pragma unroll
  for (int i = 0; i < 8; ++i) {
    const int mA = m0 + i + 8 * lhi, mB = mA + 16;
    const int nA = n0 + llo, nB = nA + 16;
    if (mA < GN && nA < GN)
      adj[(size_t)mA * GN + nA] = 1.0f / (1.0f + __expf(-acc00[i]));
    if (mA < GN && nB < GN)
      adj[(size_t)mA * GN + nB] = 1.0f / (1.0f + __expf(-acc01[i]));
    if (mB < GN && nA < GN)
      adj[(size_t)mB * GN + nA] = 1.0f / (1.0f + __expf(-acc10[i]));
    if (mB < GN && nB < GN)
      adj[(size_t)mB * GN + nB] = 1.0f / (1.0f + __expf(-acc11[i]));
  }
}

// ---------------------------------------------------------------------------
// element-wise helpers
// ---------------------------------------------------------------------------
__global__ void zero_kernel(float* __restrict__ p, int n) {
  for (int i = blockIdx.x * blockDim.x + threadIdx.x; i < n;
       i += gridDim.x * blockDim.x)
    p[i] = 0.0f;
}

__global__ void deg_kernel(const int* __restrict__ dst,
                           float* __restrict__ deg, int e) {
  const int i = blockIdx.x * blockDim.x + threadIdx.x;
  if (i < e) atomicAdd(&deg[dst[i]], 1.0f);
}

__global__ void finalize_relu_kernel(const float* __restrict__ agg,
                                     const float* __restrict__ deg,
                                     float* __restrict__ out, int d, int n) {
  const int i = blockIdx.x * blockDim.x + threadIdx.x;
  if (i < n) {
    const float dd = fmaxf(deg[i / d], 1.0f);
    out[i] = fmaxf(agg[i] / dd, 0.0f);
  }
}

__global__ void finalize_z_kernel(const float* __restrict__ agg,
                                  const float* __restrict__ deg,
                                  float* __restrict__ z,
                                  _Float16* __restrict__ zh, int n) {
  const int i = blockIdx.x * blockDim.x + threadIdx.x;
  if (i < n) {
    const float dd = fmaxf(deg[i / 64], 1.0f);
    const float v = agg[i] / dd;
    z[i] = v;
    zh[i] = (_Float16)v;
  }
}

// ---------------------------------------------------------------------------
extern "C" void kernel_launch(void* const* d_in, const int* in_sizes, int n_in,
                              void* d_out, int out_size, void* d_ws,
                              size_t ws_size, hipStream_t stream) {
  (void)in_sizes; (void)n_in; (void)out_size; (void)ws_size;

  const float* X   = (const float*)d_in[0];   // [N,128]
  const float* EF  = (const float*)d_in[1];   // [E,64]
  const int*   src = (const int*)d_in[2];     // [E]
  const int*   dst = (const int*)d_in[3];     // [E]
  const float* Wn1 = (const float*)d_in[4];   // [128,128]
  const float* bn1 = (const float*)d_in[5];
  const float* We1 = (const float*)d_in[6];   // [64,128]
  const float* be1 = (const float*)d_in[7];
  const float* Wm1 = (const float*)d_in[8];   // [128,128]
  const float* bm1 = (const float*)d_in[9];
  const float* Wn2 = (const float*)d_in[10];  // [128,64]
  const float* bn2 = (const float*)d_in[11];
  const float* We2 = (const float*)d_in[12];  // [64,64]
  const float* be2 = (const float*)d_in[13];
  const float* Wm2 = (const float*)d_in[14];  // [64,64]
  const float* bm2 = (const float*)d_in[15];

  // workspace carve (~22 MB; agg arrays stay L2-resident on 192 MB L2)
  float* ws    = (float*)d_ws;
  float* h_buf = ws;                 // [N,128]
  float* agg1  = h_buf + GN * 128;   // [N,128]
  float* h1    = agg1 + GN * 128;    // [N,128]
  float* h2    = h1 + GN * 128;      // [N,64]
  float* agg2  = h2 + GN * 64;       // [N,64]
  float* deg   = agg2 + GN * 64;     // [N]  (contiguous after agg2)
  _Float16* zh = (_Float16*)(deg + GN);  // [N,64] f16

  float* z   = (float*)d_out;        // output 0: z [N,64]
  float* adj = z + GN * 64;          // output 1: adj [N,N]

  {
    int n = GN * 128;
    zero_kernel<<<(n + 255) / 256, 256, 0, stream>>>(agg1, n);
  }
  {
    int n = GN * 64 + GN;  // agg2 + deg, contiguous
    zero_kernel<<<(n + 255) / 256, 256, 0, stream>>>(agg2, n);
  }
  deg_kernel<<<(GE + 255) / 256, 256, 0, stream>>>(dst, deg, GE);

  // ---- layer 1 ----
  fc_kernel<<<dim3(GN / 16, 128 / 16), 32, 0, stream>>>(X, Wn1, bn1, h_buf,
                                                        128, 128);
  msg_kernel<128><<<GE / (16 * ET), 32, 0, stream>>>(EF, src, dst, h_buf, We1,
                                                     be1, Wm1, bm1, agg1);
  {
    int n = GN * 128;
    finalize_relu_kernel<<<(n + 255) / 256, 256, 0, stream>>>(agg1, deg, h1,
                                                              128, n);
  }

  // ---- layer 2 ----
  fc_kernel<<<dim3(GN / 16, 64 / 16), 32, 0, stream>>>(h1, Wn2, bn2, h2, 128,
                                                       64);
  msg_kernel<64><<<GE / (16 * ET), 32, 0, stream>>>(EF, src, dst, h2, We2, be2,
                                                    Wm2, bm2, agg2);
  {
    int n = GN * 64;
    finalize_z_kernel<<<(n + 255) / 256, 256, 0, stream>>>(agg2, deg, z, zh, n);
  }

  // ---- decoder: sigmoid(z z^T), f16 WMMA, 32x32 tile per wave ----
  adj_kernel<<<dim3((GN + 31) / 32, (GN + 31) / 32), 32, 0, stream>>>(zh, adj);
}
